// DualMultiHeadAttention_91182155694157
// MI455X (gfx1250) — matrix operand order repaired and checked by reading, not compiled, not run on hardware
//
#include <hip/hip_runtime.h>

#define DEVINL static __device__ __forceinline__

constexpr int BB  = 2;
constexpr int SEQ = 2048;
constexpr int NH  = 16;
constexpr int HD  = 64;
constexpr int EMB = 1024;              // NH*HD
constexpr int MTOT = BB * SEQ;         // 4096
constexpr int BH   = BB * NH;          // 32
constexpr size_t PROJ = (size_t)BB * NH * SEQ * HD;   // 4,194,304 elements

typedef __attribute__((ext_vector_type(16))) __bf16       v16bf;
typedef __attribute__((ext_vector_type(8)))  float        v8f;
typedef __attribute__((ext_vector_type(4)))  unsigned int u32x4;
typedef __attribute__((ext_vector_type(4)))  float        f32x4;

DEVINL __bf16 f2bf(float f) {
  unsigned u = __builtin_bit_cast(unsigned, f);
  unsigned r = (u + 0x7FFFu + ((u >> 16) & 1u)) >> 16;   // round-to-nearest-even
  unsigned short h = (unsigned short)r;
  return __builtin_bit_cast(__bf16, h);
}
DEVINL float bf2f(__bf16 b) {
  unsigned short h = __builtin_bit_cast(unsigned short, b);
  unsigned u = ((unsigned)h) << 16;
  return __builtin_bit_cast(float, u);
}

union FragBF { v16bf v; u32x4 q[2]; };

// CDNA5 16x16x32 bf16 WMMA fragment layouts (ISA 7.12.2):
//   A (16x32, row-major src): lane l -> row m=l&15; half g=l>>4 holds
//     K = {8g..8g+7} and {16+8g..23+8g}  -> two 16B loads at +0 and +32B.
//   B (stored [N][K], K contiguous): lane l -> col n=l&15; half g holds
//     K = {8g..8g+7} and {16+8g..23+8g}  -> two 16B loads at +0 and +32B.
DEVINL v16bf frag_from(const __bf16* p, int stride2nd) {
  FragBF f;
  f.q[0] = *(const u32x4*)(p);
  f.q[1] = *(const u32x4*)(p + stride2nd);
  return f.v;
}

DEVINL v16bf frag_A_f32(const float* p0) {   // fp32 source, convert to bf16
  f32x4 a = *(const f32x4*)(p0);
  f32x4 b = *(const f32x4*)(p0 + 4);
  f32x4 c = *(const f32x4*)(p0 + 16);
  f32x4 d = *(const f32x4*)(p0 + 20);
  v16bf r;
  r[0]=f2bf(a[0]);  r[1]=f2bf(a[1]);  r[2]=f2bf(a[2]);  r[3]=f2bf(a[3]);
  r[4]=f2bf(b[0]);  r[5]=f2bf(b[1]);  r[6]=f2bf(b[2]);  r[7]=f2bf(b[3]);
  r[8]=f2bf(c[0]);  r[9]=f2bf(c[1]);  r[10]=f2bf(c[2]); r[11]=f2bf(c[3]);
  r[12]=f2bf(d[0]); r[13]=f2bf(d[1]); r[14]=f2bf(d[2]); r[15]=f2bf(d[3]);
  return r;
}

DEVINL v8f wmma_bf16(v16bf a, v16bf b, v8f c) {
  return __builtin_amdgcn_wmma_f32_16x16x32_bf16(false, a, false, b, (short)0, c,
                                                 false, false);
}

// ---------------------------------------------------------------- pass 0
__global__ void cvt_f32_bf16(const float* __restrict__ src,
                             __bf16* __restrict__ dst, int n) {
  int i = blockIdx.x * blockDim.x + threadIdx.x;
  if (i < n) dst[i] = f2bf(src[i]);
}

// ---------------------------------------------------------------- pass 1
// 6 projections: y = X @ W^T.  64x32 macro-tile per wave; all frag addresses
// are loop-invariant base pointers + k0 immediates (no per-iter VALU chains).
// grid (MTOT/64, EMB/256, 6), block 256 (8 waves).
__global__ void proj_gemm(const __bf16* __restrict__ xq,
                          const __bf16* __restrict__ xk,
                          const __bf16* __restrict__ xv,
                          const __bf16* __restrict__ Wall,
                          __bf16* __restrict__ Pall) {
  int p = blockIdx.z;
  int sel = p % 3;
  const __bf16* X = (sel == 0) ? xq : (sel == 1) ? xk : xv;
  const __bf16* W = Wall + (size_t)p * EMB * EMB;
  __bf16* P = Pall + (size_t)p * PROJ;
  int wv = threadIdx.x >> 5;
  int m0 = blockIdx.x * 64;
  int n0 = (blockIdx.y * 8 + wv) * 32;
  int lane = threadIdx.x & 31;
  int g = lane >> 4, lm = lane & 15;

  const __bf16* pa0 = X + (size_t)(m0 + lm) * EMB + 8 * g;
  const __bf16* pa1 = pa0 + (size_t)16 * EMB;
  const __bf16* pa2 = pa0 + (size_t)32 * EMB;
  const __bf16* pa3 = pa0 + (size_t)48 * EMB;
  const __bf16* pb0 = W + (size_t)(n0 + lm) * EMB + 8 * g;
  const __bf16* pb1 = pb0 + (size_t)16 * EMB;

  v8f c[4][2] = {};
  for (int k0 = 0; k0 < EMB; k0 += 32) {
    if (k0 + 256 < EMB)
      __builtin_prefetch((const void*)(pa0 + k0 + 256), 0, 3);
    v16bf b0 = frag_from(pb0 + k0, 16);
    v16bf b1 = frag_from(pb1 + k0, 16);
    v16bf a0 = frag_from(pa0 + k0, 16);
    c[0][0] = wmma_bf16(a0, b0, c[0][0]);
    c[0][1] = wmma_bf16(a0, b1, c[0][1]);
    v16bf a1 = frag_from(pa1 + k0, 16);
    c[1][0] = wmma_bf16(a1, b0, c[1][0]);
    c[1][1] = wmma_bf16(a1, b1, c[1][1]);
    v16bf a2 = frag_from(pa2 + k0, 16);
    c[2][0] = wmma_bf16(a2, b0, c[2][0]);
    c[2][1] = wmma_bf16(a2, b1, c[2][1]);
    v16bf a3 = frag_from(pa3 + k0, 16);
    c[3][0] = wmma_bf16(a3, b0, c[3][0]);
    c[3][1] = wmma_bf16(a3, b1, c[3][1]);
  }
#pragma unroll
  for (int t = 0; t < 4; ++t)
#pragma unroll
    for (int j = 0; j < 2; ++j) {
      int nn = n0 + 16 * j + lm;
      int h = nn >> 6, d = nn & 63;
#pragma unroll
      for (int r = 0; r < 8; ++r) {
        int m = m0 + 16 * t + r + 8 * g;
        int bb = m >> 11, s = m & (SEQ - 1);
        P[(((size_t)bb * NH + h) * SEQ + s) * HD + d] = f2bf(c[t][j][r]);
      }
    }
}

// ---------------------------------------------------------------- pass 2a
// RoPE on Q/K of both attentions (half = 32: pair (d, d+32)).
__global__ void rope_kernel(const __bf16* __restrict__ Pall,
                            __bf16* __restrict__ Qr,
                            __bf16* __restrict__ Kr) {
  const long per = (long)BB * NH * SEQ * 32;
  long idx = (long)blockIdx.x * blockDim.x + threadIdx.x;
  if (idx >= 4 * per) return;
  int sel = (int)(idx / per);        // 0:Q1 1:K1 2:Q2 3:K2
  long r = idx % per;
  int d = (int)(r & 31);
  long shs = r >> 5;                 // bh*SEQ + s
  int s = (int)(shs & (SEQ - 1));
  int a = sel >> 1, isK = sel & 1;
  const __bf16* src = Pall + (size_t)(a * 3 + isK) * PROJ;
  __bf16* dst = (isK ? Kr : Qr) + (size_t)a * PROJ;
  size_t base = (size_t)shs * HD + d;
  float x1 = bf2f(src[base]);
  float x2 = bf2f(src[base + 32]);
  float inv_freq = __expf(-(float)d * 0.28782313662425575f);  // 10000^(-d/32)
  float ang = (float)s * inv_freq;
  float sn, cs;
  __sincosf(ang, &sn, &cs);
  dst[base]      = f2bf(x1 * cs - x2 * sn);
  dst[base + 32] = f2bf(x1 * sn + x2 * cs);
}

// ---------------------------------------------------------------- pass 2b
// V transpose to [a][bh][d][s] so the ctx GEMM B-operand is K-contiguous.
__global__ void vtrans_kernel(const __bf16* __restrict__ Pall,
                              __bf16* __restrict__ Vt) {
  long idx = (long)blockIdx.x * blockDim.x + threadIdx.x;
  if (idx >= (long)2 * PROJ) return;
  int s = (int)(idx & (SEQ - 1));
  long rest = idx >> 11;
  int d = (int)(rest & 63); rest >>= 6;
  int bh = (int)(rest & 31);
  int a = (int)(rest >> 5);
  const __bf16* src = Pall + (size_t)(a * 3 + 2) * PROJ;
  Vt[(size_t)a * PROJ + ((size_t)bh * HD + d) * SEQ + s] =
      src[((size_t)bh * SEQ + s) * HD + d];
}

// ---------------------------------------------------------------- pass 3
// scores = scale * Q K^T, causal lower-triangular tiles only, written to d_out.
// grid (SEQ/16, BH, 2), block 256 (8 waves).
__global__ void scores_kernel(const __bf16* __restrict__ Qr,
                              const __bf16* __restrict__ Kr,
                              float* __restrict__ wbase) {
  int a = blockIdx.z, bh = blockIdx.y, qt = blockIdx.x;
  int wv = threadIdx.x >> 5;
  const __bf16* Q = Qr + ((size_t)a * BH + bh) * SEQ * HD;
  const __bf16* K = Kr + ((size_t)a * BH + bh) * SEQ * HD;
  float* wrow = wbase + ((size_t)a * BH + bh) * (size_t)SEQ * SEQ;
  int q0 = qt * 16;
  const float scale = 0.125f;  // 1/sqrt(64)
  int lane = threadIdx.x & 31;
  int g = lane >> 4, lm = lane & 15;

  const __bf16* pq = Q + (size_t)(q0 + lm) * HD + 8 * g;
  v16bf a0 = frag_from(pq, 16);
  v16bf a1 = frag_from(pq + 32, 16);
  const __bf16* pk = K + (size_t)lm * HD + 8 * g;   // advance by kt*16 rows

  for (int kt = wv; kt <= qt; kt += 8) {
    int k0 = kt * 16;
    const __bf16* pkk = pk + (size_t)k0 * HD;
    v8f c = {};
    c = wmma_bf16(a0, frag_from(pkk, 16), c);
    c = wmma_bf16(a1, frag_from(pkk + 32, 16), c);
#pragma unroll
    for (int r = 0; r < 8; ++r) {
      int qq = q0 + r + 8 * g;
      wrow[(size_t)qq * SEQ + k0 + lm] = c[r] * scale;
    }
  }
}

// ---------------------------------------------------------------- pass 4
// In-place causal row softmax.  grid (SEQ, BH, 2), block 256.
__global__ void softmax_kernel(float* __restrict__ wbase) {
  int q = blockIdx.x, bh = blockIdx.y, a = blockIdx.z;
  float* row = wbase + (((size_t)a * BH + bh) * SEQ + q) * (size_t)SEQ;
  __shared__ float red[256];
  int tid = threadIdx.x;
  float vals[SEQ / 256];
  float mx = -1e30f;
  for (int i = 0; i < SEQ / 256; ++i) {
    int k = tid + i * 256;
    float v = (k <= q) ? row[k] : -1e30f;
    vals[i] = v;
    mx = fmaxf(mx, v);
  }
  red[tid] = mx; __syncthreads();
  for (int o = 128; o > 0; o >>= 1) {
    if (tid < o) red[tid] = fmaxf(red[tid], red[tid + o]);
    __syncthreads();
  }
  mx = red[0]; __syncthreads();
  float sum = 0.f;
  for (int i = 0; i < SEQ / 256; ++i) {
    float e = (vals[i] > -1e29f) ? __expf(vals[i] - mx) : 0.f;
    vals[i] = e;
    sum += e;
  }
  red[tid] = sum; __syncthreads();
  for (int o = 128; o > 0; o >>= 1) {
    if (tid < o) red[tid] += red[tid + o];
    __syncthreads();
  }
  float inv = 1.f / red[0];
  for (int i = 0; i < SEQ / 256; ++i) row[tid + i * 256] = vals[i] * inv;
}

// ---------------------------------------------------------------- pass 5
// ctx = w @ V  (M=SEQ, N=HD=64, K up to the causal frontier).
// One wave per 16-row q-tile computes the full 64-wide head.
// grid (SEQ/128, BH, 2), block 256 (8 waves).
__global__ void ctx_gemm(const float* __restrict__ wbase,
                         const __bf16* __restrict__ Vt,
                         float* __restrict__ ctx) {
  int a = blockIdx.z, bh = blockIdx.y;
  int wv = threadIdx.x >> 5;
  int qt = blockIdx.x * 8 + wv;
  const float* wrow = wbase + ((size_t)a * BH + bh) * (size_t)SEQ * SEQ;
  const __bf16* V = Vt + ((size_t)a * BH + bh) * (size_t)HD * SEQ;
  float* out = ctx + ((size_t)a * BH + bh) * (size_t)SEQ * HD;
  int q0 = qt * 16;
  int kmax = q0 + 16;  // entries above the diagonal are exact zeros in w
  int lane = threadIdx.x & 31;
  int g = lane >> 4, lm = lane & 15;

  const float* pw = wrow + (size_t)(q0 + lm) * SEQ + 8 * g;
  const __bf16* pv0 = V + (size_t)lm * SEQ + 8 * g;
  const __bf16* pv1 = pv0 + (size_t)16 * SEQ;
  const __bf16* pv2 = pv0 + (size_t)32 * SEQ;
  const __bf16* pv3 = pv0 + (size_t)48 * SEQ;

  v8f c0 = {}, c1 = {}, c2 = {}, c3 = {};
  for (int k0 = 0; k0 < kmax; k0 += 32) {
    v16bf af = frag_A_f32(pw + k0);
    c0 = wmma_bf16(af, frag_from(pv0 + k0, 16), c0);
    c1 = wmma_bf16(af, frag_from(pv1 + k0, 16), c1);
    c2 = wmma_bf16(af, frag_from(pv2 + k0, 16), c2);
    c3 = wmma_bf16(af, frag_from(pv3 + k0, 16), c3);
  }
#pragma unroll
  for (int r = 0; r < 8; ++r) {
    size_t ro = (size_t)(q0 + r + 8 * g) * HD + lm;
    out[ro +  0] = c0[r];
    out[ro + 16] = c1[r];
    out[ro + 32] = c2[r];
    out[ro + 48] = c3[r];
  }
}

// ---------------------------------------------------------------- pass 6
// ctx1 * ctx2, repacked [b][s][h*64+d] as bf16.
__global__ void mulpack_kernel(const float* __restrict__ c0,
                               const float* __restrict__ c1,
                               __bf16* __restrict__ ctxp) {
  long idx = (long)blockIdx.x * blockDim.x + threadIdx.x;
  if (idx >= (long)MTOT * EMB) return;
  int e = (int)(idx & (EMB - 1));
  long bs = idx >> 10;
  int s = (int)(bs & (SEQ - 1));
  int b = (int)(bs >> 11);
  int h = e >> 6, d = e & 63;
  size_t src = (((size_t)b * NH + h) * SEQ + s) * HD + d;
  ctxp[idx] = f2bf(c0[src] * c1[src]);
}

// ---------------------------------------------------------------- pass 7
// out = ctxp @ WO^T -> fp32 d_out.  Same 64x32 macro-tile as proj_gemm.
// grid (MTOT/64, EMB/256), block 256.
__global__ void out_gemm(const __bf16* __restrict__ ctxp,
                         const __bf16* __restrict__ WO,
                         float* __restrict__ out) {
  int wv = threadIdx.x >> 5;
  int m0 = blockIdx.x * 64;
  int n0 = (blockIdx.y * 8 + wv) * 32;
  int lane = threadIdx.x & 31;
  int g = lane >> 4, lm = lane & 15;

  const __bf16* pa0 = ctxp + (size_t)(m0 + lm) * EMB + 8 * g;
  const __bf16* pa1 = pa0 + (size_t)16 * EMB;
  const __bf16* pa2 = pa0 + (size_t)32 * EMB;
  const __bf16* pa3 = pa0 + (size_t)48 * EMB;
  const __bf16* pb0 = WO + (size_t)(n0 + lm) * EMB + 8 * g;
  const __bf16* pb1 = pb0 + (size_t)16 * EMB;

  v8f c[4][2] = {};
  for (int k0 = 0; k0 < EMB; k0 += 32) {
    if (k0 + 256 < EMB)
      __builtin_prefetch((const void*)(pa0 + k0 + 256), 0, 3);
    v16bf b0 = frag_from(pb0 + k0, 16);
    v16bf b1 = frag_from(pb1 + k0, 16);
    v16bf a0 = frag_from(pa0 + k0, 16);
    c[0][0] = wmma_bf16(a0, b0, c[0][0]);
    c[0][1] = wmma_bf16(a0, b1, c[0][1]);
    v16bf a1 = frag_from(pa1 + k0, 16);
    c[1][0] = wmma_bf16(a1, b0, c[1][0]);
    c[1][1] = wmma_bf16(a1, b1, c[1][1]);
    v16bf a2 = frag_from(pa2 + k0, 16);
    c[2][0] = wmma_bf16(a2, b0, c[2][0]);
    c[2][1] = wmma_bf16(a2, b1, c[2][1]);
    v16bf a3 = frag_from(pa3 + k0, 16);
    c[3][0] = wmma_bf16(a3, b0, c[3][0]);
    c[3][1] = wmma_bf16(a3, b1, c[3][1]);
  }
#pragma unroll
  for (int t = 0; t < 4; ++t)
#pragma unroll
    for (int r = 0; r < 8; ++r) {
      size_t ro = (size_t)(m0 + 16 * t + r + 8 * g) * EMB + n0 + lm;
      out[ro]      = c[t][0][r];
      out[ro + 16] = c[t][1][r];
    }
}

// ================================================================ launch
extern "C" void kernel_launch(void* const* d_in, const int* in_sizes, int n_in,
                              void* d_out, int out_size, void* d_ws, size_t ws_size,
                              hipStream_t stream) {
  (void)in_sizes; (void)n_in; (void)out_size; (void)ws_size;

  char* ws = (char*)d_ws;
  size_t off = 0;
  auto carve = [&](size_t bytes) -> void* {
    void* p = ws + off;
    off += (bytes + 255) & ~(size_t)255;
    return p;
  };

  __bf16* xbf[3];
  for (int i = 0; i < 3; ++i) xbf[i] = (__bf16*)carve((size_t)MTOT * EMB * 2);
  __bf16* Wall = (__bf16*)carve((size_t)7 * EMB * EMB * 2);
  __bf16* Pall = (__bf16*)carve((size_t)6 * PROJ * 2);
  __bf16* Qr   = (__bf16*)carve((size_t)2 * PROJ * 2);
  __bf16* Kr   = (__bf16*)carve((size_t)2 * PROJ * 2);
  __bf16* Vt   = (__bf16*)carve((size_t)2 * PROJ * 2);
  float*  ctxf = (float*)carve((size_t)2 * PROJ * 4);
  __bf16* ctxp = (__bf16*)carve((size_t)MTOT * EMB * 2);

  float* out_f = (float*)d_out;
  float* wbase = out_f + (size_t)MTOT * EMB;   // w1 then w2, contiguous

  // pass 0: fp32 -> bf16 (3 activations + 7 weights; d_in[3] = mask, unused: causal)
  const int NE = MTOT * EMB;
  for (int i = 0; i < 3; ++i)
    cvt_f32_bf16<<<NE / 256, 256, 0, stream>>>((const float*)d_in[i], xbf[i], NE);
  for (int p = 0; p < 7; ++p)
    cvt_f32_bf16<<<(EMB * EMB) / 256, 256, 0, stream>>>(
        (const float*)d_in[4 + p], Wall + (size_t)p * EMB * EMB, EMB * EMB);

  // pass 1: 6 projection GEMMs
  proj_gemm<<<dim3(MTOT / 64, EMB / 256, 6), 256, 0, stream>>>(
      xbf[0], xbf[1], xbf[2], Wall, Pall);

  // pass 2: RoPE(Q,K) + V transpose
  rope_kernel<<<(unsigned)((4 * (PROJ / 2)) / 256), 256, 0, stream>>>(Pall, Qr, Kr);
  vtrans_kernel<<<(unsigned)((2 * PROJ) / 256), 256, 0, stream>>>(Pall, Vt);

  // pass 3: causal score tiles -> d_out w region
  scores_kernel<<<dim3(SEQ / 16, BH, 2), 256, 0, stream>>>(Qr, Kr, wbase);

  // pass 4: in-place row softmax (writes zeros above the diagonal)
  softmax_kernel<<<dim3(SEQ, BH, 2), 256, 0, stream>>>(wbase);

  // pass 5: ctx = w @ V
  ctx_gemm<<<dim3(SEQ / 128, BH, 2), 256, 0, stream>>>(wbase, Vt, ctxf);

  // pass 6: ctx1 * ctx2 -> bf16 packed [b][s][e]
  mulpack_kernel<<<(unsigned)(((size_t)MTOT * EMB) / 256), 256, 0, stream>>>(
      ctxf, ctxf + PROJ, ctxp);

  // pass 7: output projection -> d_out
  out_gemm<<<dim3(MTOT / 64, EMB / 256), 256, 0, stream>>>(
      ctxp, Wall + (size_t)6 * EMB * EMB, out_f);
}
